// CausalSelfAttention_32049045962845
// MI455X (gfx1250) — compile-verified, matrix operand order checked by reading, not run on hardware
//
#include <hip/hip_runtime.h>
#include <hip/hip_bf16.h>
#include <math.h>

// Problem constants (reference: B=4, T=2048, C=1024, H=16, d=64)
#define BN   4
#define TQ   2048
#define CDIM 1024
#define HN   16
#define DH   64
#define LDT  40   // LDS row stride (halfs) for 32-wide K tiles, keeps 16B alignment
#define LDP  40   // LDS row stride for P-transpose tiles

typedef __attribute__((ext_vector_type(16))) __bf16 v16bf;
typedef __attribute__((ext_vector_type(8)))  float  v8f;
typedef __attribute__((ext_vector_type(4)))  int    i4;

// CDNA5 async global->LDS copy path (ASYNCcnt), guarded so we always compile.
#if defined(__gfx1250__) && __has_builtin(__builtin_amdgcn_global_load_async_to_lds_b128)
#define ASYNC_OK 1
#endif

static __device__ __forceinline__ void async_cp16(__bf16* dst, const __bf16* src) {
#ifdef ASYNC_OK
  // Builtin expects (int4 AS1*, int4 AS3*, imm offset, imm cpol).
  __builtin_amdgcn_global_load_async_to_lds_b128(
      (__attribute__((address_space(1))) i4*)src,
      (__attribute__((address_space(3))) i4*)dst, 0, 0);
#else
  *(uint4*)dst = *(const uint4*)src;
#endif
}

template <int N>
static __device__ __forceinline__ void wait_async() {
#ifdef ASYNC_OK
#if __has_builtin(__builtin_amdgcn_s_wait_asynccnt)
  __builtin_amdgcn_s_wait_asynccnt(N);
#else
  asm volatile("s_wait_asynccnt %0" ::"i"(N) : "memory");
#endif
#endif
}

static __device__ __forceinline__ v8f wmma_bf16(v16bf a, v16bf b, v8f c) {
  // D = A(16x32 bf16) * B(32x16 bf16) + C(16x16 f32)
  return __builtin_amdgcn_wmma_f32_16x16x32_bf16(false, a, false, b, (short)0, c,
                                                 false, false);
}

// Load a 16x32 bf16 A/B fragment from row-major storage with leading dim `ld`.
// ISA layout: lane L<16 -> row L, K in {0..7, 16..23}; lane L>=16 -> row L-16,
// K in {8..15, 24..31}.  Both 8-half groups are contiguous -> b128 loads.
static __device__ __forceinline__ v16bf load_frag16(const __bf16* base, int ld) {
  const int lane = threadIdx.x & 31;
  const int r = lane & 15;
  const int kb = (lane >> 4) << 3;
  const __bf16* p = base + r * ld + kb;
  v16bf f;
#pragma unroll
  for (int e = 0; e < 8; ++e) { f[e] = p[e]; f[e + 8] = p[e + 16]; }
  return f;
}

// Reductions across the 16-lane halves of a wave32 (C/D layout: lane = column).
static __device__ __forceinline__ float redmax16(float v) {
#pragma unroll
  for (int m = 1; m < 16; m <<= 1) v = fmaxf(v, __shfl_xor(v, m, 32));
  return v;
}
static __device__ __forceinline__ float redsum16(float v) {
#pragma unroll
  for (int m = 1; m < 16; m <<= 1) v += __shfl_xor(v, m, 32);
  return v;
}

// ---------------------------------------------------------------------------
// Kernel 0: zero the attention-mean output region (upper triangle must be 0).
__global__ void zero_f32(float4* __restrict__ p, int n4) {
  int i = blockIdx.x * blockDim.x + threadIdx.x;
  int stride = gridDim.x * blockDim.x;
  for (; i < n4; i += stride) p[i] = make_float4(0.f, 0.f, 0.f, 0.f);
}

// Kernel 0b: fp32 -> bf16 bulk conversion (8 elements / thread / iter).
__global__ void cvt_bf16(const float* __restrict__ in, __bf16* __restrict__ outp,
                         int n8) {
  int i = blockIdx.x * blockDim.x + threadIdx.x;
  int stride = gridDim.x * blockDim.x;
  for (; i < n8; i += stride) {
    const float4 v0 = ((const float4*)in)[2 * (size_t)i];
    const float4 v1 = ((const float4*)in)[2 * (size_t)i + 1];
    __align__(16) __bf16 o[8] = {(__bf16)v0.x, (__bf16)v0.y, (__bf16)v0.z,
                                 (__bf16)v0.w, (__bf16)v1.x, (__bf16)v1.y,
                                 (__bf16)v1.z, (__bf16)v1.w};
    *(uint4*)(outp + 8 * (size_t)i) = *(const uint4*)o;
  }
}

// ---------------------------------------------------------------------------
// Kernel 1: qkv = x @ w_attn^T (M=8192, N=3072, K=1024), bf16 WMMA.
// 128x64 block tile, 4 waves, each wave 32x64 (8 WMMA / K-step).
// Double-buffered async global->LDS staging.
// q,k stored [B,H,T,d] bf16; v stored transposed [B,H,d,T] bf16.
__global__ __launch_bounds__(128) void qkv_gemm(const __bf16* __restrict__ xb,
                                                const __bf16* __restrict__ wab,
                                                __bf16* __restrict__ qb,
                                                __bf16* __restrict__ kbuf,
                                                __bf16* __restrict__ vT) {
  __shared__ __align__(16) __bf16 As[2][128 * LDT];
  __shared__ __align__(16) __bf16 Bs[2][64 * LDT];
  const int tid = threadIdx.x;
  const int lane = tid & 31, wv = tid >> 5;
  const int Mb = blockIdx.x * 128, Nb = blockIdx.y * 64;

  auto stage = [&](int buf, int k0) {
#pragma unroll
    for (int c = tid; c < 512; c += 128) {  // A: 128x32 halfs = 512 x 16B
      int row = c >> 2, c8 = (c & 3) << 3;
      async_cp16(&As[buf][row * LDT + c8],
                 xb + (size_t)(Mb + row) * CDIM + k0 + c8);
    }
#pragma unroll
    for (int c = tid; c < 256; c += 128) {  // B: 64x32 halfs = 256 x 16B
      int row = c >> 2, c8 = (c & 3) << 3;
      async_cp16(&Bs[buf][row * LDT + c8],
                 wab + (size_t)(Nb + row) * CDIM + k0 + c8);
    }
  };

  v8f acc[2][4] = {};
  stage(0, 0);
  for (int t = 0; t < CDIM / 32; ++t) {
    const int cur = t & 1;
    if (t + 1 < CDIM / 32) {
      stage(cur ^ 1, (t + 1) * 32);
      wait_async<6>();   // 6 async copies/wave/tile outstanding for next tile
    } else {
      wait_async<0>();
    }
    __syncthreads();
    const v16bf a0 = load_frag16(&As[cur][(wv * 32) * LDT], LDT);
    const v16bf a1 = load_frag16(&As[cur][(wv * 32 + 16) * LDT], LDT);
#pragma unroll
    for (int nb = 0; nb < 4; ++nb) {
      const v16bf bf = load_frag16(&Bs[cur][nb * 16 * LDT], LDT);
      acc[0][nb] = wmma_bf16(a0, bf, acc[0][nb]);
      acc[1][nb] = wmma_bf16(a1, bf, acc[1][nb]);
    }
    __syncthreads();
  }

  // Scatter to q / k / v^T (bf16).  C/D layout: lane holds column n, rows mo+r.
  const int n = lane & 15, mo = (lane >> 4) << 3;
#pragma unroll
  for (int a = 0; a < 2; ++a) {
#pragma unroll
    for (int nb = 0; nb < 4; ++nb) {
#pragma unroll
      for (int r = 0; r < 8; ++r) {
        const int trow = Mb + wv * 32 + a * 16 + mo + r;
        const int b = trow >> 11, t = trow & (TQ - 1);
        const int o = Nb + nb * 16 + n;
        const int part = o >> 10, oo = o & (CDIM - 1);
        const int h = oo >> 6, dd = oo & (DH - 1);
        const __bf16 val = (__bf16)acc[a][nb][r];
        const size_t bh = (size_t)(b * HN + h);
        if (part == 0)      qb[(bh * TQ + t) * DH + dd] = val;
        else if (part == 1) kbuf[(bh * TQ + t) * DH + dd] = val;
        else                vT[(bh * DH + dd) * TQ + t] = val;
      }
    }
  }
}

// ---------------------------------------------------------------------------
// Kernel 2a: online softmax stats (row max m, denom l) per (b,h,q-tile).
// Lane-LOCAL online update (no shuffles in the k-loop); one cross-lane merge
// at the end using the flash-attention combine rule.
__global__ __launch_bounds__(128) void attn_stats(const __bf16* __restrict__ qb,
                                                  const __bf16* __restrict__ kbuf,
                                                  float* __restrict__ sm,
                                                  float* __restrict__ sl) {
  const int lane = threadIdx.x & 31, wv = threadIdx.x >> 5;
  const int qt = (blockIdx.x & 31) * 4 + wv;   // 128 q-tiles per (b,h)
  const int bh = blockIdx.x >> 5;              // 0..B*H-1
  const int qbase = qt * 16;
  const int n = lane & 15, mo = (lane >> 4) << 3;

  const __bf16* qp = qb + ((size_t)bh * TQ + qbase) * DH;
  const __bf16* kp = kbuf + (size_t)bh * TQ * DH;
  const v16bf aq0 = load_frag16(qp, DH);
  const v16bf aq1 = load_frag16(qp + 32, DH);

  float mlane[8], llane[8];
#pragma unroll
  for (int r = 0; r < 8; ++r) { mlane[r] = -INFINITY; llane[r] = 0.f; }

  for (int j = 0; j <= qt; ++j) {
    const __bf16* kj = kp + (size_t)j * 16 * DH;
    v8f s = {};
    s = wmma_bf16(aq0, load_frag16(kj, DH), s);
    s = wmma_bf16(aq1, load_frag16(kj + 32, DH), s);
    const bool diag = (j == qt);
#pragma unroll
    for (int r = 0; r < 8; ++r) {
      const float sv = s[r] * 0.125f;           // 1/sqrt(64)
      if (!(diag && n > mo + r)) {              // skip masked: avoid inf-inf NaN
        const float mn = fmaxf(mlane[r], sv);
        llane[r] = llane[r] * __expf(mlane[r] - mn) + __expf(sv - mn);
        mlane[r] = mn;
      }
    }
  }
  // Cross-lane merge: m = max over columns, l = sum l_i * exp(m_i - m).
#pragma unroll
  for (int r = 0; r < 8; ++r) {
    const float m = redmax16(mlane[r]);
    const float l = redsum16(llane[r] * __expf(mlane[r] - m));
    if (n == 0) {
      const size_t idx = (size_t)bh * TQ + qbase + mo + r;
      sm[idx] = m;
      sl[idx] = l;
    }
  }
}

// ---------------------------------------------------------------------------
// Kernel 2b: per (h,q-tile), loop batches in registers: recompute S, normalize
// with (m,l), accumulate batch-mean attention tile (plain stores, no atomics),
// transpose P through LDS into A-fragment layout, accumulate O = P @ V (WMMA).
__global__ __launch_bounds__(128) void attn_out(const __bf16* __restrict__ qb,
                                                const __bf16* __restrict__ kbuf,
                                                const __bf16* __restrict__ vT,
                                                const float* __restrict__ sm,
                                                const float* __restrict__ sl,
                                                float* __restrict__ attn,
                                                __bf16* __restrict__ oh) {
  __shared__ __align__(16) __bf16 pt[4][BN][16 * LDP];
  const int lane = threadIdx.x & 31, wv = threadIdx.x >> 5;
  const int h  = blockIdx.x >> 5;
  const int qt = (blockIdx.x & 31) * 4 + wv;
  const int qbase = qt * 16;
  const int n = lane & 15, mo = (lane >> 4) << 3;

  v16bf aq[BN][2];
  float mS[BN][8], il[BN][8];
#pragma unroll
  for (int b = 0; b < BN; ++b) {
    const __bf16* qp = qb + (((size_t)(b * HN + h)) * TQ + qbase) * DH;
    aq[b][0] = load_frag16(qp, DH);
    aq[b][1] = load_frag16(qp + 32, DH);
#pragma unroll
    for (int r = 0; r < 8; ++r) {
      const size_t idx = (size_t)(b * HN + h) * TQ + qbase + mo + r;
      mS[b][r] = sm[idx];
      il[b][r] = 1.f / sl[idx];
    }
  }

  v8f o[BN][4] = {};
  for (int j = 0; j <= qt; ++j) {
    const bool diag = (j == qt);
    const int colb = (j & 1) << 4;
    v8f mt = {};
#pragma unroll
    for (int b = 0; b < BN; ++b) {
      const __bf16* kj = kbuf + (((size_t)(b * HN + h)) * TQ + (size_t)j * 16) * DH;
      v8f s = {};
      s = wmma_bf16(aq[b][0], load_frag16(kj, DH), s);
      s = wmma_bf16(aq[b][1], load_frag16(kj + 32, DH), s);
      __bf16* ptb = &pt[wv][b][0];
#pragma unroll
      for (int r = 0; r < 8; ++r) {
        float p;
        if (diag && n > mo + r) p = 0.f;
        else p = __expf(s[r] * 0.125f - mS[b][r]) * il[b][r];
        mt[r] += 0.25f * p;                               // mean over B=4
        ptb[(mo + r) * LDP + colb + n] = (__bf16)p;       // P[m][kcol] for A-frag
      }
    }
    // Batch-mean attention tile -> global (plain stores)
#pragma unroll
    for (int r = 0; r < 8; ++r)
      attn[((size_t)h * TQ + qbase + mo + r) * TQ + (size_t)j * 16 + n] = mt[r];

    // Flush a 32-wide K chunk of P @ V
    if ((j & 1) == 1 || diag) {
      const int tbase = (j & ~1) * 16;
#pragma unroll
      for (int b = 0; b < BN; ++b) {
        __bf16* ptb = &pt[wv][b][0];
        if ((j & 1) == 0) {
          // chunk under-filled (odd tile count): zero columns 16..31
          for (int i = lane; i < 256; i += 32)
            ptb[(i >> 4) * LDP + 16 + (i & 15)] = (__bf16)0.f;
        }
        const v16bf pf = load_frag16(ptb, LDP);
        const __bf16* vp = vT + ((size_t)(b * HN + h)) * DH * TQ + tbase;
#pragma unroll
        for (int nb = 0; nb < 4; ++nb) {
          v16bf bv = load_frag16(vp + (size_t)nb * 16 * TQ, TQ);  // contiguous in t
          o[b][nb] = wmma_bf16(pf, bv, o[b][nb]);
        }
      }
    }
  }

  // Write head outputs as bf16 in [B, T, H*d] layout for the projection GEMM.
#pragma unroll
  for (int b = 0; b < BN; ++b)
#pragma unroll
    for (int nb = 0; nb < 4; ++nb)
#pragma unroll
      for (int r = 0; r < 8; ++r)
        oh[((size_t)b * TQ + qbase + mo + r) * CDIM + h * DH + nb * 16 + n] =
            (__bf16)o[b][nb][r];
}

// ---------------------------------------------------------------------------
// Kernel 3: out = oh @ w_proj^T (M=8192, N=1024, K=1024), fp32 output.
// Same double-buffered async staging + 32x64 wave tiles.
__global__ __launch_bounds__(128) void proj_gemm(const __bf16* __restrict__ oh,
                                                 const __bf16* __restrict__ wpb,
                                                 float* __restrict__ out) {
  __shared__ __align__(16) __bf16 As[2][128 * LDT];
  __shared__ __align__(16) __bf16 Bs[2][64 * LDT];
  const int tid = threadIdx.x;
  const int lane = tid & 31, wv = tid >> 5;
  const int Mb = blockIdx.x * 128, Nb = blockIdx.y * 64;

  auto stage = [&](int buf, int k0) {
#pragma unroll
    for (int c = tid; c < 512; c += 128) {
      int row = c >> 2, c8 = (c & 3) << 3;
      async_cp16(&As[buf][row * LDT + c8],
                 oh + (size_t)(Mb + row) * CDIM + k0 + c8);
    }
#pragma unroll
    for (int c = tid; c < 256; c += 128) {
      int row = c >> 2, c8 = (c & 3) << 3;
      async_cp16(&Bs[buf][row * LDT + c8],
                 wpb + (size_t)(Nb + row) * CDIM + k0 + c8);
    }
  };

  v8f acc[2][4] = {};
  stage(0, 0);
  for (int t = 0; t < CDIM / 32; ++t) {
    const int cur = t & 1;
    if (t + 1 < CDIM / 32) {
      stage(cur ^ 1, (t + 1) * 32);
      wait_async<6>();
    } else {
      wait_async<0>();
    }
    __syncthreads();
    const v16bf a0 = load_frag16(&As[cur][(wv * 32) * LDT], LDT);
    const v16bf a1 = load_frag16(&As[cur][(wv * 32 + 16) * LDT], LDT);
#pragma unroll
    for (int nb = 0; nb < 4; ++nb) {
      const v16bf bf = load_frag16(&Bs[cur][nb * 16 * LDT], LDT);
      acc[0][nb] = wmma_bf16(a0, bf, acc[0][nb]);
      acc[1][nb] = wmma_bf16(a1, bf, acc[1][nb]);
    }
    __syncthreads();
  }

  const int n = lane & 15, mo = (lane >> 4) << 3;
#pragma unroll
  for (int a = 0; a < 2; ++a)
#pragma unroll
    for (int nb = 0; nb < 4; ++nb)
#pragma unroll
      for (int r = 0; r < 8; ++r)
        out[(size_t)(Mb + wv * 32 + a * 16 + mo + r) * CDIM + Nb + nb * 16 + n] =
            acc[a][nb][r];
}

// ---------------------------------------------------------------------------
extern "C" void kernel_launch(void* const* d_in, const int* in_sizes, int n_in,
                              void* d_out, int out_size, void* d_ws, size_t ws_size,
                              hipStream_t stream) {
  (void)in_sizes; (void)n_in; (void)out_size; (void)ws_size;
  const float* x     = (const float*)d_in[0];   // [B,T,C]
  const float* wattn = (const float*)d_in[1];   // [3C,C]
  const float* wproj = (const float*)d_in[2];   // [C,C]

  float* out  = (float*)d_out;                              // [B,T,C]
  float* attn = out + (size_t)BN * TQ * CDIM;               // [H,T,T]

  // Workspace layout (bf16 operand copies + q/k/vT/oh + fp32 stats): ~93 MB
  const size_t NE = (size_t)BN * TQ * CDIM;                 // 8,388,608 elements
  __bf16* xb  = (__bf16*)d_ws;                              // [B*T, C]
  __bf16* wab = xb + NE;                                    // [3C, C]
  __bf16* wpb = wab + (size_t)3 * CDIM * CDIM;              // [C, C]
  __bf16* qb  = wpb + (size_t)CDIM * CDIM;                  // [B,H,T,d]
  __bf16* kb  = qb + NE;
  __bf16* vT  = kb + NE;                                    // [B,H,d,T]
  __bf16* oh  = vT + NE;                                    // [B,T,C]
  float*  sm  = (float*)(oh + NE);                          // [B,H,T]
  float*  sl  = sm + (size_t)BN * HN * TQ;

  // 0) zero attention-mean region (masked entries must be 0)
  zero_f32<<<4096, 256, 0, stream>>>((float4*)attn, (HN * TQ * TQ) / 4);
  // 0b) fp32 -> bf16 operand conversions
  cvt_bf16<<<2048, 256, 0, stream>>>(x, xb, (int)(NE / 8));
  cvt_bf16<<<1024, 256, 0, stream>>>(wattn, wab, 3 * CDIM * CDIM / 8);
  cvt_bf16<<<512, 256, 0, stream>>>(wproj, wpb, CDIM * CDIM / 8);
  // 1) qkv GEMM + repack (v transposed)
  qkv_gemm<<<dim3(64, 48), 128, 0, stream>>>(xb, wab, qb, kb, vT);
  // 2a) online-softmax row stats per (b,h,q-tile)
  attn_stats<<<BN * HN * 32, 128, 0, stream>>>(qb, kb, sm, sl);
  // 2b) attention weights (batch mean) + P@V head outputs
  attn_out<<<HN * 32, 128, 0, stream>>>(qb, kb, vT, sm, sl, attn, oh);
  // 3) output projection
  proj_gemm<<<dim3(64, 16), 128, 0, stream>>>(oh, wpb, out);
}
